// Attention_47622597378403
// MI455X (gfx1250) — matrix operand order repaired and checked by reading, not compiled, not run on hardware
//
#include <hip/hip_runtime.h>

#define BQ 2
#define LQ 2048
#define DIMQ 2048
#define NH 16
#define NKV 4
#define HD 128
#define WIN 512
#define SCALE 0.08838834764831845f   // 1/sqrt(128)

typedef __bf16 bf16;
typedef __attribute__((ext_vector_type(16))) __bf16 v16bf;
typedef __attribute__((ext_vector_type(8)))  float  v8f;
typedef __attribute__((ext_vector_type(4)))  unsigned int u32x4;

union FragU { v16bf v; u32x4 q[2]; };

#define WMMA_BF16(a, b, c) \
  __builtin_amdgcn_wmma_f32_16x16x32_bf16(false, (a), false, (b), (short)0, (c), false, false)

// A-fragment: 16x32 bf16, row-major source, tile top-left at src, row stride ld.
// ISA layout: lanes 0-15 row m=lane, K={0..7,16..23}; lanes 16-31 row m=lane-16, K={8..15,24..31}.
static __device__ inline v16bf load_a_frag(const bf16* src, int ld, int lane) {
  int r    = lane & 15;
  int koff = (lane < 16) ? 0 : 8;
  FragU f;
  f.q[0] = *(const u32x4*)(src + (size_t)r * ld + koff);
  f.q[1] = *(const u32x4*)(src + (size_t)r * ld + 16 + koff);
  return f.v;
}

// B-fragment: 32x16 bf16 fed from B^T stored row-major [N, K] with row stride ld.
// ISA layout: lane n holds column n of B; lanes 0-15: K={0..7,16..23}, lanes 16-31: K={8..15,24..31}.
static __device__ inline v16bf load_b_frag(const bf16* srcT, int ld, int lane) {
  int n    = lane & 15;
  int koff = (lane < 16) ? 0 : 8;
  const bf16* p = srcT + (size_t)n * ld + koff;
  FragU f;
  f.q[0] = *(const u32x4*)(p);
  f.q[1] = *(const u32x4*)(p + 16);
  return f.v;
}

// ---------------- conversion / packing ----------------

__global__ void cvt_f32_to_bf16(const float* __restrict__ in, bf16* __restrict__ out, int n) {
  for (int i = blockIdx.x * blockDim.x + threadIdx.x; i < n; i += gridDim.x * blockDim.x)
    out[i] = (bf16)in[i];
}

// in: [rows(K), cols(N)] fp32 row-major  ->  outT: [cols(N), rows(K)] bf16 row-major
__global__ void transpose_to_bf16(const float* __restrict__ in, bf16* __restrict__ outT,
                                  int rows, int cols) {
  int total = rows * cols;
  for (int i = blockIdx.x * blockDim.x + threadIdx.x; i < total; i += gridDim.x * blockDim.x) {
    int c = i / rows;
    int r = i - c * rows;
    outT[i] = (bf16)in[(size_t)r * cols + c];
  }
}

// ---------------- GEMM core (one wave = 16x64 tile), software-pipelined k-loop --------------

static __device__ inline void gemm_tile_16x64(const bf16* __restrict__ Arow,
                                              const bf16* __restrict__ BTtile, // + colG*64*K
                                              int K, int lane, v8f c[4]) {
  v16bf a_cur = load_a_frag(Arow, K, lane);
  v16bf b_cur0 = load_b_frag(BTtile + (size_t)0  * K, K, lane);
  v16bf b_cur1 = load_b_frag(BTtile + (size_t)16 * K, K, lane);
  v16bf b_cur2 = load_b_frag(BTtile + (size_t)32 * K, K, lane);
  v16bf b_cur3 = load_b_frag(BTtile + (size_t)48 * K, K, lane);
  for (int k0 = 0; k0 < K - 32; k0 += 32) {
    int k1 = k0 + 32;
    // issue next-step loads before consuming current fragments
    v16bf a_nxt  = load_a_frag(Arow + k1, K, lane);
    v16bf b_nxt0 = load_b_frag(BTtile + (size_t)0  * K + k1, K, lane);
    v16bf b_nxt1 = load_b_frag(BTtile + (size_t)16 * K + k1, K, lane);
    v16bf b_nxt2 = load_b_frag(BTtile + (size_t)32 * K + k1, K, lane);
    v16bf b_nxt3 = load_b_frag(BTtile + (size_t)48 * K + k1, K, lane);
    c[0] = WMMA_BF16(a_cur, b_cur0, c[0]);
    c[1] = WMMA_BF16(a_cur, b_cur1, c[1]);
    c[2] = WMMA_BF16(a_cur, b_cur2, c[2]);
    c[3] = WMMA_BF16(a_cur, b_cur3, c[3]);
    a_cur = a_nxt;
    b_cur0 = b_nxt0; b_cur1 = b_nxt1; b_cur2 = b_nxt2; b_cur3 = b_nxt3;
  }
  c[0] = WMMA_BF16(a_cur, b_cur0, c[0]);
  c[1] = WMMA_BF16(a_cur, b_cur1, c[1]);
  c[2] = WMMA_BF16(a_cur, b_cur2, c[2]);
  c[3] = WMMA_BF16(a_cur, b_cur3, c[3]);
}

// C[M,N] = A[M,K] @ B[K,N], A bf16 row-major, B supplied as BT [N,K], C stored bf16 row-major.
__global__ __launch_bounds__(256) void gemm_rm_bf16(const bf16* __restrict__ A,
                                                    const bf16* __restrict__ BT,
                                                    bf16* __restrict__ C,
                                                    int M, int N, int K) {
  int wave = (blockIdx.x * blockDim.x + threadIdx.x) >> 5;
  int lane = threadIdx.x & 31;
  int colGroups = N >> 6;
  if (wave >= (M >> 4) * colGroups) return;
  int rowTile = wave / colGroups;
  int colG    = wave % colGroups;
  v8f c[4] = {};
  gemm_tile_16x64(A + (size_t)rowTile * 16 * K, BT + (size_t)colG * 64 * K, K, lane, c);
  int n  = lane & 15;
  int mb = (lane < 16) ? 0 : 8;
#pragma unroll
  for (int t = 0; t < 4; ++t) {
    int col = colG * 64 + t * 16 + n;
#pragma unroll
    for (int r = 0; r < 8; ++r)
      C[(size_t)(rowTile * 16 + mb + r) * N + col] = (bf16)c[t][r];
  }
}

// Same compute, but stores V transposed: Vt[b][hkv][d][L]  (N must be NKV*HD=512).
__global__ __launch_bounds__(256) void gemm_vt_bf16(const bf16* __restrict__ A,
                                                    const bf16* __restrict__ BT,
                                                    bf16* __restrict__ Vt,
                                                    int M, int N, int K) {
  int wave = (blockIdx.x * blockDim.x + threadIdx.x) >> 5;
  int lane = threadIdx.x & 31;
  int colGroups = N >> 6;
  if (wave >= (M >> 4) * colGroups) return;
  int rowTile = wave / colGroups;
  int colG    = wave % colGroups;
  v8f c[4] = {};
  gemm_tile_16x64(A + (size_t)rowTile * 16 * K, BT + (size_t)colG * 64 * K, K, lane, c);
  int n  = lane & 15;
  int mb = (lane < 16) ? 0 : 8;
#pragma unroll
  for (int t = 0; t < 4; ++t) {
    int col = colG * 64 + t * 16 + n;   // col = hkv*128 + d
    int hk  = col >> 7;
    int d   = col & 127;
#pragma unroll
    for (int r = 0; r < 8; ++r) {
      int m   = rowTile * 16 + mb + r;  // m = b*L + token
      int b_  = m / LQ;
      int tok = m - b_ * LQ;
      Vt[((size_t)(b_ * NKV + hk) * HD + d) * LQ + tok] = (bf16)c[t][r];
    }
  }
}

// Output projection: fp32 store.
__global__ __launch_bounds__(256) void gemm_out_f32(const bf16* __restrict__ A,
                                                    const bf16* __restrict__ BT,
                                                    float* __restrict__ C,
                                                    int M, int N, int K) {
  int wave = (blockIdx.x * blockDim.x + threadIdx.x) >> 5;
  int lane = threadIdx.x & 31;
  int colGroups = N >> 6;
  if (wave >= (M >> 4) * colGroups) return;
  int rowTile = wave / colGroups;
  int colG    = wave % colGroups;
  v8f c[4] = {};
  gemm_tile_16x64(A + (size_t)rowTile * 16 * K, BT + (size_t)colG * 64 * K, K, lane, c);
  int n  = lane & 15;
  int mb = (lane < 16) ? 0 : 8;
#pragma unroll
  for (int t = 0; t < 4; ++t) {
    int col = colG * 64 + t * 16 + n;
#pragma unroll
    for (int r = 0; r < 8; ++r)
      C[(size_t)(rowTile * 16 + mb + r) * N + col] = c[t][r];
  }
}

// ---------------- flash attention (windowed, GQA) ----------------
// One wave = one (b, h, 16-query tile). 4 waves/block share (b, qtile) and one KV head
// (heads 4g..4g+3 -> hkv=g), so loop trip counts are block-uniform (barrier-safe).

__global__ __launch_bounds__(128) void attn_kernel(const bf16* __restrict__ Qb,
                                                   const bf16* __restrict__ Kb,
                                                   const bf16* __restrict__ Vt,
                                                   bf16* __restrict__ Ab) {
  __shared__ __align__(16) float Sld[4][16][33];
  __shared__ __align__(16) bf16  Pld[4][16][32];
  __shared__ __align__(16) float alphaLd[4][16];

  int lane = threadIdx.x & 31;
  int wv   = threadIdx.x >> 5;            // 0..3
  int bid  = blockIdx.x;
  int qTile = bid % (LQ / 16);
  int rest  = bid / (LQ / 16);
  int hg    = rest % (NH / 4);            // head group == kv head
  int b_    = rest / (NH / 4);
  int h     = hg * 4 + wv;
  int hk    = hg;
  int q0    = qTile * 16;

  const bf16* Qtile  = Qb + ((size_t)(b_ * LQ + q0)) * (NH * HD) + (size_t)h * HD; // ld = 2048
  const bf16* Kbase  = Kb + ((size_t)(b_ * LQ)) * (NKV * HD) + (size_t)hk * HD;    // ld = 512
  const bf16* Vtbase = Vt + ((size_t)(b_ * NKV + hk)) * HD * LQ;                   // ld = L

  v8f o[8] = {};
  float m_i = -__builtin_inff();
  float l_i = 0.f;

  int n  = lane & 15;
  int mb = (lane < 16) ? 0 : 8;

  int klo = q0 - WIN;
  if (klo < 0) klo = 0;
  klo &= ~31;

  for (int kstart = klo; kstart <= q0 + 15; kstart += 32) {
    const bf16* K0 = Kbase + (size_t)kstart * (NKV * HD);
    const bf16* K1 = Kbase + (size_t)(kstart + 16) * (NKV * HD);

    // ---- S = Q K^T (16 queries x 32 keys), pipelined over head_dim chunks ----
    v8f s0 = {}, s1 = {};
    v16bf aq = load_a_frag(Qtile, NH * HD, lane);
    v16bf bk0 = load_b_frag(K0, NKV * HD, lane);
    v16bf bk1 = load_b_frag(K1, NKV * HD, lane);
#pragma unroll
    for (int kc = 0; kc < HD - 32; kc += 32) {
      v16bf aq_n  = load_a_frag(Qtile + kc + 32, NH * HD, lane);
      v16bf bk0_n = load_b_frag(K0 + kc + 32, NKV * HD, lane);
      v16bf bk1_n = load_b_frag(K1 + kc + 32, NKV * HD, lane);
      s0 = WMMA_BF16(aq, bk0, s0);
      s1 = WMMA_BF16(aq, bk1, s1);
      aq = aq_n; bk0 = bk0_n; bk1 = bk1_n;
    }
    s0 = WMMA_BF16(aq, bk0, s0);
    s1 = WMMA_BF16(aq, bk1, s1);

#pragma unroll
    for (int r = 0; r < 8; ++r) {
      Sld[wv][mb + r][n]      = s0[r] * SCALE;
      Sld[wv][mb + r][16 + n] = s1[r] * SCALE;
    }

    // prefetch next key block's K rows while softmax/PV runs (global_prefetch_b8)
    int knext = kstart + 32;
    if (knext <= q0 + 15)
      __builtin_prefetch(Kbase + (size_t)(knext + lane) * (NKV * HD), 0, 3);

    __syncthreads();

    // ---- online softmax: lane r owns query row q0+r ----
    if (lane < 16) {
      int q = q0 + lane;
      float mloc = -__builtin_inff();
#pragma unroll
      for (int j = 0; j < 32; ++j) {
        int k = kstart + j;
        bool ok = (k <= q) && (q - k <= WIN);
        float s = ok ? Sld[wv][lane][j] : -__builtin_inff();
        Sld[wv][lane][j] = s;                 // write masked value back
        mloc = fmaxf(mloc, s);
      }
      float mnew = fmaxf(m_i, mloc);
      float alpha;
      if (mnew == -__builtin_inff()) {        // entire block masked so far
        alpha = 1.f;
#pragma unroll
        for (int j = 0; j < 32; ++j) Pld[wv][lane][j] = (bf16)0.0f;
      } else {
        alpha = __expf(m_i - mnew);           // m_i == -inf -> 0
        float rowsum = 0.f;
#pragma unroll
        for (int j = 0; j < 32; ++j) {
          float s = Sld[wv][lane][j];
          float p = (s == -__builtin_inff()) ? 0.f : __expf(s - mnew);
          rowsum += p;
          Pld[wv][lane][j] = (bf16)p;
        }
        l_i = l_i * alpha + rowsum;
        m_i = mnew;
      }
      alphaLd[wv][lane] = alpha;
    }
    __syncthreads();

    // ---- rescale accumulator, then O += P @ V ----
    float al[8];
#pragma unroll
    for (int r = 0; r < 8; ++r) al[r] = alphaLd[wv][mb + r];
#pragma unroll
    for (int t = 0; t < 8; ++t)
#pragma unroll
      for (int r = 0; r < 8; ++r) o[t][r] *= al[r];

    v16bf ap = load_a_frag(&Pld[wv][0][0], 32, lane);
#pragma unroll
    for (int t = 0; t < 8; ++t) {
      v16bf bv = load_b_frag(Vtbase + (size_t)(16 * t) * LQ + kstart, LQ, lane);
      o[t] = WMMA_BF16(ap, bv, o[t]);
    }
    __syncthreads();
  }

  // ---- finalize: divide by row sums (0 for fully-masked rows, like nan_to_num) ----
  if (lane < 16) alphaLd[wv][lane] = (l_i > 0.f) ? (1.f / l_i) : 0.f;
  __syncthreads();
  float sc[8];
#pragma unroll
  for (int r = 0; r < 8; ++r) sc[r] = alphaLd[wv][mb + r];

#pragma unroll
  for (int t = 0; t < 8; ++t)
#pragma unroll
    for (int r = 0; r < 8; ++r) {
      int q = q0 + mb + r;
      Ab[((size_t)(b_ * LQ + q)) * (NH * HD) + (size_t)h * HD + 16 * t + n] =
          (bf16)(o[t][r] * sc[r]);
    }
}

// ---------------- host launcher ----------------

extern "C" void kernel_launch(void* const* d_in, const int* in_sizes, int n_in,
                              void* d_out, int out_size, void* d_ws, size_t ws_size,
                              hipStream_t stream) {
  (void)in_sizes; (void)n_in; (void)out_size; (void)ws_size;
  const float* x  = (const float*)d_in[0];
  const float* wq = (const float*)d_in[1];
  const float* wk = (const float*)d_in[2];
  const float* wv = (const float*)d_in[3];
  const float* wo = (const float*)d_in[4];
  float* out = (float*)d_out;

  char* ws = (char*)d_ws;
  size_t off = 0;
  auto alloc = [&](size_t bytes) -> char* {
    char* p = ws + off;
    off += (bytes + 255) & ~(size_t)255;
    return p;
  };

  const int M = BQ * LQ;  // 4096 tokens
  bf16* xb  = (bf16*)alloc((size_t)M * DIMQ * 2);
  bf16* wqT = (bf16*)alloc((size_t)DIMQ * (NH * HD) * 2);
  bf16* wkT = (bf16*)alloc((size_t)DIMQ * (NKV * HD) * 2);
  bf16* wvT = (bf16*)alloc((size_t)DIMQ * (NKV * HD) * 2);
  bf16* woT = (bf16*)alloc((size_t)(NH * HD) * DIMQ * 2);
  bf16* Qb  = (bf16*)alloc((size_t)M * (NH * HD) * 2);
  bf16* Kb  = (bf16*)alloc((size_t)M * (NKV * HD) * 2);
  bf16* Vt  = (bf16*)alloc((size_t)BQ * NKV * HD * LQ * 2);
  bf16* Ab  = (bf16*)alloc((size_t)M * (NH * HD) * 2);

  // 1) pack inputs to bf16 (weights transposed to [N,K] for B-fragment loads)
  cvt_f32_to_bf16<<<4096, 256, 0, stream>>>(x, xb, M * DIMQ);
  transpose_to_bf16<<<4096, 256, 0, stream>>>(wq, wqT, DIMQ, NH * HD);
  transpose_to_bf16<<<1024, 256, 0, stream>>>(wk, wkT, DIMQ, NKV * HD);
  transpose_to_bf16<<<1024, 256, 0, stream>>>(wv, wvT, DIMQ, NKV * HD);
  transpose_to_bf16<<<4096, 256, 0, stream>>>(wo, woT, NH * HD, DIMQ);

  // 2) projections:  Q [M,2048], K [M,512] row-major bf16;  V stored transposed
  gemm_rm_bf16<<<1024, 256, 0, stream>>>(xb, wqT, Qb, M, NH * HD, DIMQ);
  gemm_rm_bf16<<<256, 256, 0, stream>>>(xb, wkT, Kb, M, NKV * HD, DIMQ);
  gemm_vt_bf16<<<256, 256, 0, stream>>>(xb, wvT, Vt, M, NKV * HD, DIMQ);

  // 3) windowed GQA flash attention
  attn_kernel<<<BQ * (LQ / 16) * (NH / 4), 128, 0, stream>>>(Qb, Kb, Vt, Ab);

  // 4) output projection -> fp32
  gemm_out_f32<<<1024, 256, 0, stream>>>(Ab, woT, out, M, DIMQ, NH * HD);
}
